// ConvFlow_14216341749988
// MI455X (gfx1250) — matrix operand order repaired and checked by reading, not compiled, not run on hardware
//
#include <hip/hip_runtime.h>
#include <math.h>

#define B_     32
#define T_     8192
#define FILT_  192
#define KS_    3
#define NL_    3
#define NBINS_ 10
#define TAIL_  5.0f
#define MIN_W_ 0.001f
#define MIN_D_ 0.001f
#define NPAR_  29     // 3*NBINS-1
#define TT_    32     // t-tile for dwconv kernel

typedef __attribute__((ext_vector_type(16))) _Float16 v16h;
typedef __attribute__((ext_vector_type(8)))  _Float16 v8h;
typedef __attribute__((ext_vector_type(8)))  float    v8f;
typedef __attribute__((ext_vector_type(4)))  float    v4f;

__device__ __forceinline__ float gelu_exact(float v) {
  return 0.5f * v * (1.0f + erff(v * 0.70710678118654752f));
}

// ---------------------------------------------------------------------------
// K0: h[b][t][o] = pre_w[o]*x0[b][t] + pre_b[o]; also out[:,0,:] = x0
// ---------------------------------------------------------------------------
__global__ __launch_bounds__(256) void k_pre(const float* __restrict__ x,
                                             const float* __restrict__ pre_w,
                                             const float* __restrict__ pre_b,
                                             float* __restrict__ h,
                                             float* __restrict__ out) {
  const size_t total = (size_t)B_ * T_ * FILT_;
  for (size_t idx = blockIdx.x * (size_t)blockDim.x + threadIdx.x; idx < total;
       idx += (size_t)gridDim.x * blockDim.x) {
    int o = (int)(idx % FILT_);
    size_t bt = idx / FILT_;
    int t = (int)(bt % T_);
    int b = (int)(bt / T_);
    float x0 = x[((size_t)b * 2) * T_ + t];
    h[idx] = pre_w[o] * x0 + pre_b[o];
    if (idx < (size_t)B_ * T_) {
      int tb = (int)(idx % T_);
      int bb = (int)(idx / T_);
      out[((size_t)bb * 2) * T_ + tb] = x[((size_t)bb * 2) * T_ + tb];
    }
  }
}

// ---------------------------------------------------------------------------
// K1: fused dilated depthwise conv (K=3) + channel-LN + exact GELU -> f16
// h layout [B][T][C] fp32 ; yt layout [B][T][C] f16
// ---------------------------------------------------------------------------
__global__ __launch_bounds__(256) void k_dwln(const float* __restrict__ h,
                                              _Float16* __restrict__ yt,
                                              const float* __restrict__ dw_w,
                                              const float* __restrict__ dw_b,
                                              const float* __restrict__ g1,
                                              const float* __restrict__ b1,
                                              int layer, int dil) {
  extern __shared__ char smem[];
  float* hbuf = (float*)smem;            // up to (TT_+18)*192 = 38400 B
  float* yloc = (float*)(smem + 38400);  // 32*192*4 = 24576 B
  float* ps   = (float*)(smem + 62976);  // 8*32*4
  float* pq   = (float*)(smem + 64000);  // 8*32*4
  float* muv  = (float*)(smem + 65024);  // 32*4
  float* rsv  = (float*)(smem + 65152);  // 32*4

  const int tid   = threadIdx.x;
  const int chunk = blockIdx.x % (T_ / TT_);
  const int b     = blockIdx.x / (T_ / TT_);
  const int t0    = chunk * TT_;
  const int nrows = TT_ + 2 * dil;

  const float* hb = h + ((size_t)b * T_) * FILT_;
  for (int e = tid; e < nrows * FILT_; e += blockDim.x) {
    int r = e / FILT_, c = e % FILT_;
    int gt = t0 + r - dil;
    hbuf[r * FILT_ + c] = (gt >= 0 && gt < T_) ? hb[(size_t)gt * FILT_ + c] : 0.0f;
  }
  __syncthreads();

  const float* wdw = dw_w + (size_t)layer * FILT_ * KS_;
  const float* bdw = dw_b + (size_t)layer * FILT_;
  for (int e = tid; e < TT_ * FILT_; e += blockDim.x) {
    int t = e / FILT_, c = e % FILT_;
    float a = bdw[c];
    a += wdw[c * 3 + 0] * hbuf[(t + 0 * dil) * FILT_ + c];
    a += wdw[c * 3 + 1] * hbuf[(t + 1 * dil) * FILT_ + c];
    a += wdw[c * 3 + 2] * hbuf[(t + 2 * dil) * FILT_ + c];
    yloc[e] = a;
  }
  __syncthreads();

  { // LayerNorm over 192 channels: 8 partial groups of 24 per t-position
    int t = tid & 31, j = tid >> 5;
    float s = 0.f, q = 0.f;
    for (int k = 0; k < 24; ++k) {
      float v = yloc[t * FILT_ + j * 24 + k];
      s += v; q += v * v;
    }
    ps[j * 32 + t] = s; pq[j * 32 + t] = q;
  }
  __syncthreads();
  if (tid < 32) {
    float s = 0.f, q = 0.f;
    for (int j = 0; j < 8; ++j) { s += ps[j * 32 + tid]; q += pq[j * 32 + tid]; }
    float mu = s * (1.0f / FILT_);
    float var = q * (1.0f / FILT_) - mu * mu;
    muv[tid] = mu;
    rsv[tid] = rsqrtf(var + 1e-5f);
  }
  __syncthreads();

  const float* gg = g1 + (size_t)layer * FILT_;
  const float* bb = b1 + (size_t)layer * FILT_;
  _Float16* yo = yt + ((size_t)b * T_ + t0) * FILT_;
  for (int e = tid; e < TT_ * FILT_; e += blockDim.x) {
    int t = e / FILT_, c = e % FILT_;
    float v = (yloc[e] - muv[t]) * rsv[t] * gg[c] + bb[c];
    yo[e] = (_Float16)gelu_exact(v);
  }
}

// ---------------------------------------------------------------------------
// K2: WMMA pointwise 192x192 GEMM + bias + channel-LN + GELU + residual
// Each wave: one 16-column (t) block, 12 M-tiles x 6 K-steps of
// v_wmma_f32_16x16x32_f16. A/B both use contiguous K = 16*hi + s per lane.
// ---------------------------------------------------------------------------
__global__ __launch_bounds__(256) void k_pwln(const float* __restrict__ pw_w,
                                              const float* __restrict__ pw_b,
                                              const float* __restrict__ g2,
                                              const float* __restrict__ b2,
                                              const _Float16* __restrict__ yt,
                                              float* __restrict__ h,
                                              int layer) {
  extern __shared__ char smem[];
  _Float16* wl = (_Float16*)smem;          // 192*192 f16 = 73728 B
  float* l2g = (float*)(smem + 73728);     // 192
  float* l2b = (float*)(smem + 74496);     // 192
  float* pb  = (float*)(smem + 75264);     // 192

  const int tid = threadIdx.x;
  const float* wsrc = pw_w + (size_t)layer * FILT_ * FILT_;
  for (int e = tid; e < FILT_ * FILT_; e += blockDim.x) wl[e] = (_Float16)wsrc[e];
  for (int e = tid; e < FILT_; e += blockDim.x) {
    l2g[e] = g2[layer * FILT_ + e];
    l2b[e] = b2[layer * FILT_ + e];
    pb[e]  = pw_b[layer * FILT_ + e];
  }
  __syncthreads();

  const int b    = blockIdx.x / (T_ / 128);
  const int t0   = (blockIdx.x % (T_ / 128)) * 128;
  const int wave = tid >> 5;
  const int lane = tid & 31;
  const int n    = lane & 15;   // A-row within M-tile == B-column (t) index
  const int hi   = lane >> 4;   // lane-half selector for K ranges

  const int t = t0 + wave * 16 + n;
  const _Float16* yrow = yt + ((size_t)b * T_ + t) * FILT_;

  v8f acc[12];
#pragma unroll
  for (int mt = 0; mt < 12; ++mt) { v8f z = {}; acc[mt] = z; }

#pragma unroll
  for (int kc = 0; kc < 6; ++kc) {
    const int k0 = kc * 32;
    // B 32x16 f16: lane holds col n, K = 16*hi .. 16*hi+15 (contiguous)
    v8h bl0 = *(const v8h*)(yrow + k0 + 16 * hi);
    v8h bl1 = *(const v8h*)(yrow + k0 + 16 * hi + 8);
    v16h bv = __builtin_shufflevector(bl0, bl1, 0,1,2,3,4,5,6,7,8,9,10,11,12,13,14,15);
#pragma unroll
    for (int mt = 0; mt < 12; ++mt) {
      // A 16x32 f16: lane holds row m=n; K = 16*hi .. 16*hi+15 (contiguous, matches B)
      const _Float16* wrow = wl + (mt * 16 + n) * FILT_ + k0;
      v8h a0 = *(const v8h*)(wrow + 16 * hi);
      v8h a1 = *(const v8h*)(wrow + 16 * hi + 8);
      v16h av = __builtin_shufflevector(a0, a1, 0,1,2,3,4,5,6,7,8,9,10,11,12,13,14,15);
      acc[mt] = __builtin_amdgcn_wmma_f32_16x16x32_f16(
          false, av, false, bv, (short)0, acc[mt], false, false);
    }
  }

  // Epilogue: bias + LN over 192 outputs of column t. D layout: lane holds
  // rows m = mt*16 + r + 8*hi; lane and lane^16 share column -> one shfl_xor.
  float s = 0.f, q = 0.f;
#pragma unroll
  for (int mt = 0; mt < 12; ++mt)
#pragma unroll
    for (int r = 0; r < 8; ++r) {
      int mm = mt * 16 + r + 8 * hi;
      float v = acc[mt][r] + pb[mm];
      s += v; q += v * v;
    }
  s += __shfl_xor(s, 16, 32);
  q += __shfl_xor(q, 16, 32);
  float mu  = s * (1.0f / FILT_);
  float var = q * (1.0f / FILT_) - mu * mu;
  float rs  = rsqrtf(var + 1e-5f);

  float* hrow = h + ((size_t)b * T_ + t) * FILT_;
#pragma unroll
  for (int mt = 0; mt < 12; ++mt)
#pragma unroll
    for (int r = 0; r < 8; ++r) {
      int mm = mt * 16 + r + 8 * hi;
      float v = (acc[mt][r] + pb[mm] - mu) * rs * l2g[mm] + l2b[mm];
      hrow[mm] += gelu_exact(v);
    }
}

// ---------------------------------------------------------------------------
// K3: 29x192 projection + rational-quadratic spline + partial logdet sums
// ---------------------------------------------------------------------------
__global__ __launch_bounds__(256) void k_spline(const float* __restrict__ x,
                                                const float* __restrict__ proj_w,
                                                const float* __restrict__ proj_b,
                                                const float* __restrict__ h,
                                                float* __restrict__ out,
                                                float* __restrict__ partial) {
  __shared__ float pw[NPAR_ * FILT_];
  __shared__ float pbs[NPAR_];
  __shared__ float red[256];
  const int tid = threadIdx.x;
  for (int e = tid; e < NPAR_ * FILT_; e += blockDim.x) pw[e] = proj_w[e];
  if (tid < NPAR_) pbs[tid] = proj_b[tid];
  __syncthreads();

  const int b = blockIdx.x / (T_ / 256);
  const int t = (blockIdx.x % (T_ / 256)) * 256 + tid;
  const float* hrow = h + ((size_t)b * T_ + t) * FILT_;

  float acc[NPAR_];
#pragma unroll
  for (int o = 0; o < NPAR_; ++o) acc[o] = pbs[o];
  for (int c = 0; c < FILT_; c += 4) {
    v4f hv = *(const v4f*)(hrow + c);
#pragma unroll
    for (int o = 0; o < NPAR_; ++o)
      acc[o] += pw[o * FILT_ + c + 0] * hv[0] + pw[o * FILT_ + c + 1] * hv[1] +
                pw[o * FILT_ + c + 2] * hv[2] + pw[o * FILT_ + c + 3] * hv[3];
  }

  const float scale = 0.07216878364870323f;  // 1/sqrt(192)
  float cw[NBINS_ + 1], ch[NBINS_ + 1], wd[NBINS_], hg[NBINS_], dv[NBINS_ + 1];

  { // bin widths
    float m = -1e30f;
#pragma unroll
    for (int j = 0; j < NBINS_; ++j) m = fmaxf(m, acc[j] * scale);
    float sum = 0.f;
#pragma unroll
    for (int j = 0; j < NBINS_; ++j) { wd[j] = expf(acc[j] * scale - m); sum += wd[j]; }
    float inv = 1.0f / sum, cum = 0.f;
    cw[0] = -TAIL_;
#pragma unroll
    for (int j = 0; j < NBINS_; ++j) {
      cum += MIN_W_ + (1.0f - MIN_W_ * NBINS_) * wd[j] * inv;
      cw[j + 1] = 2.0f * TAIL_ * cum - TAIL_;
    }
    cw[NBINS_] = TAIL_;
#pragma unroll
    for (int j = 0; j < NBINS_; ++j) wd[j] = cw[j + 1] - cw[j];
  }
  { // bin heights
    float m = -1e30f;
#pragma unroll
    for (int j = 0; j < NBINS_; ++j) m = fmaxf(m, acc[NBINS_ + j] * scale);
    float sum = 0.f;
#pragma unroll
    for (int j = 0; j < NBINS_; ++j) { hg[j] = expf(acc[NBINS_ + j] * scale - m); sum += hg[j]; }
    float inv = 1.0f / sum, cum = 0.f;
    ch[0] = -TAIL_;
#pragma unroll
    for (int j = 0; j < NBINS_; ++j) {
      cum += MIN_W_ + (1.0f - MIN_W_ * NBINS_) * hg[j] * inv;  // MIN_H == MIN_W
      ch[j + 1] = 2.0f * TAIL_ * cum - TAIL_;
    }
    ch[NBINS_] = TAIL_;
#pragma unroll
    for (int j = 0; j < NBINS_; ++j) hg[j] = ch[j + 1] - ch[j];
  }
  // derivatives: boundary pad value gives exactly MIN_D + softplus(log(e^{1-MIN_D}-1)) = 1
  dv[0] = 1.0f; dv[NBINS_] = 1.0f;
#pragma unroll
  for (int j = 0; j < NBINS_ - 1; ++j) {
    float u = acc[2 * NBINS_ + j];
    float sp = (u > 20.0f) ? u : log1pf(expf(u));
    dv[j + 1] = MIN_D_ + sp;
  }

  float x1v = x[((size_t)b * 2 + 1) * T_ + t];
  bool inside = (x1v >= -TAIL_) && (x1v <= TAIL_);
  float xin = fminf(fmaxf(x1v, -TAIL_), TAIL_);
  int cnt = 0;
#pragma unroll
  for (int j = 0; j <= NBINS_; ++j) {
    float loc = cw[j] + ((j == NBINS_) ? 1e-6f : 0.0f);
    cnt += (xin >= loc) ? 1 : 0;
  }
  int idx = cnt - 1;
  idx = idx < 0 ? 0 : (idx > NBINS_ - 1 ? NBINS_ - 1 : idx);

  float icw = cw[idx], iw = wd[idx], ich = ch[idx], ih = hg[idx];
  float delta = ih / iw;
  float d0 = dv[idx], d1 = dv[idx + 1];
  float th  = (xin - icw) / iw;
  float th1 = th * (1.0f - th);
  float num = ih * (delta * th * th + d0 * th1);
  float den = delta + (d0 + d1 - 2.0f * delta) * th1;
  float xo  = inside ? (ich + num / den) : x1v;
  float omt = 1.0f - th;
  float dnum = delta * delta * (d1 * th * th + 2.0f * delta * th1 + d0 * omt * omt);
  float lad = inside ? (logf(dnum) - 2.0f * logf(den)) : 0.0f;

  out[((size_t)b * 2 + 1) * T_ + t] = xo;

  red[tid] = lad;
  __syncthreads();
  for (int off = 128; off > 0; off >>= 1) {
    if (tid < off) red[tid] += red[tid + off];
    __syncthreads();
  }
  if (tid == 0) partial[blockIdx.x] = red[0];
}

// ---------------------------------------------------------------------------
// K4: deterministic per-batch reduction of logdet partials
// ---------------------------------------------------------------------------
__global__ void k_reduce(const float* __restrict__ partial, float* __restrict__ ld) {
  int b = blockIdx.x;
  float v = partial[b * (T_ / 256) + threadIdx.x];  // 32 partials per batch
  for (int off = 16; off > 0; off >>= 1) v += __shfl_down(v, off, 32);
  if (threadIdx.x == 0) ld[b] = v;
}

// ---------------------------------------------------------------------------
extern "C" void kernel_launch(void* const* d_in, const int* in_sizes, int n_in,
                              void* d_out, int out_size, void* d_ws, size_t ws_size,
                              hipStream_t stream) {
  const float* x      = (const float*)d_in[0];
  const float* pre_w  = (const float*)d_in[1];
  const float* pre_b  = (const float*)d_in[2];
  const float* dw_w   = (const float*)d_in[3];
  const float* dw_b   = (const float*)d_in[4];
  const float* ln1_g  = (const float*)d_in[5];
  const float* ln1_b  = (const float*)d_in[6];
  const float* pw_w   = (const float*)d_in[7];
  const float* pw_b   = (const float*)d_in[8];
  const float* ln2_g  = (const float*)d_in[9];
  const float* ln2_b  = (const float*)d_in[10];
  const float* proj_w = (const float*)d_in[11];
  const float* proj_b = (const float*)d_in[12];

  float* out = (float*)d_out;
  const size_t H_BYTES = (size_t)B_ * T_ * FILT_ * sizeof(float);     // 192 MB
  const size_t Y_BYTES = (size_t)B_ * T_ * FILT_ * sizeof(_Float16);  // 96 MB
  float*    h       = (float*)d_ws;
  _Float16* yt      = (_Float16*)((char*)d_ws + H_BYTES);
  float*    partial = (float*)((char*)d_ws + H_BYTES + Y_BYTES);

  k_pre<<<4096, 256, 0, stream>>>(x, pre_w, pre_b, h, out);

  int dil = 1;
  for (int i = 0; i < NL_; ++i) {
    k_dwln<<<B_ * (T_ / TT_), 256, 65280, stream>>>(h, yt, dw_w, dw_b, ln1_g, ln1_b, i, dil);
    k_pwln<<<B_ * (T_ / 128), 256, 76032, stream>>>(pw_w, pw_b, ln2_g, ln2_b, yt, h, i);
    dil *= 3;
  }

  k_spline<<<B_ * (T_ / 256), 256, 0, stream>>>(x, proj_w, proj_b, h, out, partial);
  k_reduce<<<B_, 32, 0, stream>>>(partial, out + (size_t)B_ * 2 * T_);
}
